// HybridCTM_45861660787086
// MI455X (gfx1250) — matrix operand order repaired and checked by reading, not compiled, not run on hardware
//
#include <hip/hip_runtime.h>

// Problem dimensions (match reference)
#define B_    4096
#define IN_   1024
#define H_    1024
#define S_    64
#define Q_    16
#define OUT_  1024
#define CIN_  (IN_ + H_)    // 2048
#define OGK_  (2 * H_)      // 2048
#define OGN_  (H_ + OUT_)   // 2048

typedef __bf16 bf16_t;
typedef __bf16        v16bf __attribute__((ext_vector_type(16)));
typedef float         v8f   __attribute__((ext_vector_type(8)));
typedef unsigned int  u32x4 __attribute__((ext_vector_type(4)));
typedef int           v4i_g __attribute__((vector_size(16)));   // matches builtin param pointee

union FragCvt { u32x4 q[2]; v16bf v; };

// gfx1250 async global->LDS path (ASYNCcnt-tracked DMA, no VGPR staging)
#if defined(__HIP_DEVICE_COMPILE__) && defined(__gfx1250__) &&                  \
    __has_builtin(__builtin_amdgcn_global_load_async_to_lds_b128) &&            \
    __has_builtin(__builtin_amdgcn_s_wait_asynccnt)
#define HAVE_ASYNC_LDS 1
#define ASYNC_GPTR(p) ((__attribute__((address_space(1))) v4i_g*)(p))
#define ASYNC_LPTR(p) ((__attribute__((address_space(3))) v4i_g*)(p))
#else
#define HAVE_ASYNC_LDS 0
#endif

// ---------------------------------------------------------------------------
// Kernel 1: concat(x_t, h_prev) and convert to bf16 (B x 2048)
// ---------------------------------------------------------------------------
__global__ __launch_bounds__(256) void concat_bf16_kernel(
    const float* __restrict__ x, const float* __restrict__ h,
    bf16_t* __restrict__ out)
{
    size_t i = (size_t)blockIdx.x * 256 + threadIdx.x;   // exactly B_*CIN_ threads
    size_t row = i / CIN_;
    int    col = (int)(i % CIN_);
    float v = (col < IN_) ? x[row * IN_ + col] : h[row * H_ + (col - IN_)];
    out[i] = (bf16_t)v;
}

// ---------------------------------------------------------------------------
// Kernel 2: W (K x N, f32, row-major) -> Wt (N x K, bf16, row-major)
// ---------------------------------------------------------------------------
__global__ __launch_bounds__(256) void convert_transpose_kernel(
    const float* __restrict__ W, bf16_t* __restrict__ Wt, int K, int N)
{
    __shared__ float tile[32][33];
    int n0 = blockIdx.x * 32, k0 = blockIdx.y * 32;
    int tx = threadIdx.x & 31, ty = threadIdx.x >> 5;    // 32 x 8
    for (int i = ty; i < 32; i += 8)
        tile[i][tx] = W[(size_t)(k0 + i) * N + (n0 + tx)];
    __syncthreads();
    for (int i = ty; i < 32; i += 8)
        Wt[(size_t)(n0 + i) * K + (k0 + tx)] = (bf16_t)tile[tx][i];
}

// ---------------------------------------------------------------------------
// Kernel 3: tiled bf16 WMMA GEMM:  C(MxN) = A(MxK) * Bt(NxK)^T + bias
// Double-buffered LDS; async global->LDS DMA when available.
// Epilogue modes:
//   0: outF = v                           (f32)
//   1: outF = tanh(v); outB = bf16(tanh)  (f32 + bf16)
//   2: outB = bf16(v)
//   3: n < splitN -> outF = tanh(v) ; else outF2[n-splitN] = v
// ---------------------------------------------------------------------------
#define BK 32
#define LDS_STRIDE (BK + 8)   // +16B pad per row; row pitch 80B stays 16B aligned

template <int BM, int BN, int WAVES_M, int WAVES_N>
__global__ __launch_bounds__(256) void gemm_wmma_kernel(
    const bf16_t* __restrict__ A, int lda,
    const bf16_t* __restrict__ Bt,               // N x K, ld = K
    const float*  __restrict__ bias,
    int M, int N, int K,
    int mode,
    float* __restrict__ outF, int ldF,
    bf16_t* __restrict__ outB, int ldB,
    float* __restrict__ outF2, int splitN)
{
    constexpr int WTM = BM / WAVES_M;
    constexpr int WTN = BN / WAVES_N;
    constexpr int TM  = WTM / 16;
    constexpr int TN  = WTN / 16;
    constexpr int A_ITERS = (BM * (BK / 8)) / 256;   // 16B chunks per thread
    constexpr int B_ITERS = (BN * (BK / 8)) / 256;

    __shared__ bf16_t sA[2][BM * LDS_STRIDE];
    __shared__ bf16_t sB[2][BN * LDS_STRIDE];

    const int tid   = threadIdx.x;
    const int lane  = tid & 31;
    const int wave  = tid >> 5;
    const int wm    = wave % WAVES_M;
    const int wn    = wave / WAVES_M;
    const int lmod  = lane & 15;
    const int lhalf = lane >> 4;

    const int bm = blockIdx.y * BM;
    const int bn = blockIdx.x * BN;

    // issue the staging loads for K-tile at k0 into LDS buffer `buf`
    auto loadTiles = [&](int k0, int buf) {
#pragma unroll
        for (int it = 0; it < A_ITERS; ++it) {
            int c = tid + it * 256;
            int row = c >> 2, kc = c & 3;
            const bf16_t* g = &A[(size_t)(bm + row) * lda + k0 + kc * 8];
            bf16_t*       l = &sA[buf][row * LDS_STRIDE + kc * 8];
#if HAVE_ASYNC_LDS
            __builtin_amdgcn_global_load_async_to_lds_b128(
                ASYNC_GPTR(g), ASYNC_LPTR(l), 0, 0);
#else
            *(u32x4*)l = *(const u32x4*)g;
#endif
        }
#pragma unroll
        for (int it = 0; it < B_ITERS; ++it) {
            int c = tid + it * 256;
            int row = c >> 2, kc = c & 3;
            const bf16_t* g = &Bt[(size_t)(bn + row) * K + kc * 8 + k0];
            bf16_t*       l = &sB[buf][row * LDS_STRIDE + kc * 8];
#if HAVE_ASYNC_LDS
            __builtin_amdgcn_global_load_async_to_lds_b128(
                ASYNC_GPTR(g), ASYNC_LPTR(l), 0, 0);
#else
            *(u32x4*)l = *(const u32x4*)g;
#endif
        }
    };
    auto waitStage = [&]() {
#if HAVE_ASYNC_LDS
        __builtin_amdgcn_s_wait_asynccnt(0);
#endif
    };

    v8f acc[TM][TN];
#pragma unroll
    for (int i = 0; i < TM; ++i)
#pragma unroll
        for (int j = 0; j < TN; ++j)
#pragma unroll
            for (int e = 0; e < 8; ++e) acc[i][j][e] = 0.0f;

    // prologue: stage tile 0
    loadTiles(0, 0);
    waitStage();
    __syncthreads();

    int buf = 0;
    for (int k0 = 0; k0 < K; k0 += BK) {
        // prefetch next K-tile into the other buffer (overlaps with WMMA below)
        if (k0 + BK < K) loadTiles(k0 + BK, buf ^ 1);

        // A fragment (16-bit 16x32 layout): lanes 0-15 K{0-7,16-23}, lanes 16-31 K{8-15,24-31}
        v16bf afrag[TM];
#pragma unroll
        for (int i = 0; i < TM; ++i) {
            int m = wm * WTM + i * 16 + lmod;
            const bf16_t* p = &sA[buf][m * LDS_STRIDE + lhalf * 8];
            FragCvt f;
            f.q[0] = *(const u32x4*)p;
            f.q[1] = *(const u32x4*)(p + 16);
            afrag[i] = f.v;
        }
        // B fragment: column = lane%16, lanes 0-15 K{0-7,16-23}, lanes 16-31 K{8-15,24-31}
        v16bf bfrag[TN];
#pragma unroll
        for (int j = 0; j < TN; ++j) {
            int n = wn * WTN + j * 16 + lmod;
            const bf16_t* p = &sB[buf][n * LDS_STRIDE + lhalf * 8];
            FragCvt f;
            f.q[0] = *(const u32x4*)p;
            f.q[1] = *(const u32x4*)(p + 16);
            bfrag[j] = f.v;
        }

#pragma unroll
        for (int i = 0; i < TM; ++i)
#pragma unroll
            for (int j = 0; j < TN; ++j)
                acc[i][j] = __builtin_amdgcn_wmma_f32_16x16x32_bf16(
                    false, afrag[i], false, bfrag[j],
                    (short)0, acc[i][j], false, false);

        // my async writes landed + everyone finished reading `buf`
        waitStage();
        __syncthreads();
        buf ^= 1;
    }

    // epilogue: C/D layout -> lane row = lhalf*8 + r, col = lane%16
#pragma unroll
    for (int i = 0; i < TM; ++i) {
#pragma unroll
        for (int j = 0; j < TN; ++j) {
            int gm0 = bm + wm * WTM + i * 16 + lhalf * 8;
            int gn  = bn + wn * WTN + j * 16 + lmod;
            float bv = bias[gn];
#pragma unroll
            for (int r = 0; r < 8; ++r) {
                int gm = gm0 + r;
                float v = acc[i][j][r] + bv;
                if (mode == 0) {
                    outF[(size_t)gm * ldF + gn] = v;
                } else if (mode == 1) {
                    float t = tanhf(v);
                    outF[(size_t)gm * ldF + gn] = t;
                    outB[(size_t)gm * ldB + gn] = (bf16_t)t;
                } else if (mode == 2) {
                    outB[(size_t)gm * ldB + gn] = (bf16_t)v;
                } else { // mode 3: split h_next / y_t
                    if (gn < splitN) outF [(size_t)gm * ldF + gn]            = tanhf(v);
                    else             outF2[(size_t)gm * ldF + (gn - splitN)] = v;
                }
            }
        }
    }
}

// ---------------------------------------------------------------------------
// Kernel 4: per-row argmax(logits) -> gather Wm_enc/theta/Wm_dec
//   angles = h_int . Wi + theta ; m_t = tanh(cos(angles) . Wo + b_m)  (bf16 out)
// One 256-thread block per row.
// ---------------------------------------------------------------------------
__global__ __launch_bounds__(256) void select_einsum_kernel(
    const float* __restrict__ logits,   // B x S
    const float* __restrict__ h_int,    // B x H (f32)
    const float* __restrict__ Wm_enc,   // S x H x Q
    const float* __restrict__ theta,    // S x Q
    const float* __restrict__ Wm_dec,   // S x Q x H
    const float* __restrict__ b_m,      // H
    bf16_t* __restrict__ m_out, int ld_m)
{
    const int b = blockIdx.x;
    const int tid = threadIdx.x;

    __shared__ float s_h[H_];
    __shared__ float s_red[256];
    __shared__ float s_expz[Q_];
    __shared__ int   s_idx;

    for (int h = tid; h < H_; h += 256) s_h[h] = h_int[(size_t)b * H_ + h];

    if (tid == 0) {  // argmax(softmax(logits)) == argmax(logits), first-max wins
        const float* lr = logits + (size_t)b * S_;
        float best = lr[0]; int bi = 0;
        for (int i = 1; i < S_; ++i) { float v = lr[i]; if (v > best) { best = v; bi = i; } }
        s_idx = bi;
    }
    __syncthreads();
    const int idx = s_idx;

    // angles: thread (seg, q) does a 64-long partial dot; reduce across 16 segs
    const float* Wi = Wm_enc + (size_t)idx * H_ * Q_;
    const int q = tid & 15, seg = tid >> 4;
    float part = 0.0f;
    const int h0 = seg * (H_ / 16);
    for (int h = h0; h < h0 + H_ / 16; ++h)
        part += s_h[h] * Wi[(size_t)h * Q_ + q];
    s_red[tid] = part;
    __syncthreads();

    if (tid < Q_) {
        float a = 0.0f;
        for (int s2 = 0; s2 < 16; ++s2) a += s_red[s2 * 16 + tid];
        a += theta[(size_t)idx * Q_ + tid];
        s_expz[tid] = cosf(a);
    }
    __syncthreads();

    const float* Wo = Wm_dec + (size_t)idx * Q_ * H_;
    for (int h = tid; h < H_; h += 256) {
        float acc = b_m[h];
#pragma unroll
        for (int qq = 0; qq < Q_; ++qq) acc += s_expz[qq] * Wo[(size_t)qq * H_ + h];
        m_out[(size_t)b * ld_m + h] = (bf16_t)tanhf(acc);
    }
}

// ---------------------------------------------------------------------------
// Launch
// ---------------------------------------------------------------------------
extern "C" void kernel_launch(void* const* d_in, const int* in_sizes, int n_in,
                              void* d_out, int out_size, void* d_ws, size_t ws_size,
                              hipStream_t stream)
{
    (void)in_sizes; (void)n_in; (void)out_size; (void)ws_size;

    const float* x_t    = (const float*)d_in[0];
    const float* h_prev = (const float*)d_in[1];
    const float* W_in   = (const float*)d_in[2];
    const float* b_in   = (const float*)d_in[3];
    const float* W_q    = (const float*)d_in[4];
    const float* b_q    = (const float*)d_in[5];
    const float* W_sel  = (const float*)d_in[6];
    const float* b_sel  = (const float*)d_in[7];
    const float* Wm_enc = (const float*)d_in[8];
    const float* theta  = (const float*)d_in[9];
    const float* Wm_dec = (const float*)d_in[10];
    const float* b_m    = (const float*)d_in[11];
    const float* W_og   = (const float*)d_in[12];
    const float* b_og   = (const float*)d_in[13];

    float* out    = (float*)d_out;
    float* y_t    = out;                        // B x OUT (first output)
    float* h_next = out + (size_t)B_ * OUT_;    // B x H   (second output)

    char* p = (char*)d_ws;
    auto take = [&](size_t bytes) -> void* {
        void* r = (void*)p; p += (bytes + 255) & ~(size_t)255; return r;
    };
    bf16_t* combined = (bf16_t*)take((size_t)B_ * CIN_ * 2);   // x||h bf16
    bf16_t* WtIn     = (bf16_t*)take((size_t)CIN_ * H_ * 2);   // H  x CIN
    bf16_t* WtQ      = (bf16_t*)take((size_t)H_ * H_ * 2);     // H  x H
    bf16_t* WtSel    = (bf16_t*)take((size_t)S_ * H_ * 2);     // S  x H
    bf16_t* WtOg     = (bf16_t*)take((size_t)OGN_ * OGK_ * 2); // OGN x OGK
    float*  hintF    = (float*) take((size_t)B_ * H_ * 4);     // h_int f32
    bf16_t* combOut  = (bf16_t*)take((size_t)B_ * OGK_ * 2);   // h_int||m_t bf16
    bf16_t* queryB   = (bf16_t*)take((size_t)B_ * H_ * 2);     // query bf16
    float*  logits   = (float*) take((size_t)B_ * S_ * 4);     // logits f32

    // 1) activations -> bf16
    concat_bf16_kernel<<<(B_ * CIN_) / 256, 256, 0, stream>>>(x_t, h_prev, combined);

    // 2) weights -> transposed bf16 (N x K)
    convert_transpose_kernel<<<dim3(H_  / 32, CIN_ / 32), 256, 0, stream>>>(W_in,  WtIn,  CIN_, H_);
    convert_transpose_kernel<<<dim3(H_  / 32, H_   / 32), 256, 0, stream>>>(W_q,   WtQ,   H_,   H_);
    convert_transpose_kernel<<<dim3(S_  / 32, H_   / 32), 256, 0, stream>>>(W_sel, WtSel, H_,   S_);
    convert_transpose_kernel<<<dim3(OGN_/ 32, OGK_ / 32), 256, 0, stream>>>(W_og,  WtOg,  OGK_, OGN_);

    // 3) h_int = tanh(combined @ W_in + b_in)   (f32 + bf16 into combOut[:, :H])
    gemm_wmma_kernel<128, 128, 4, 2><<<dim3(H_ / 128, B_ / 128), 256, 0, stream>>>(
        combined, CIN_, WtIn, b_in, B_, H_, CIN_,
        1, hintF, H_, combOut, OGK_, nullptr, 0);

    // 4) query = h_int @ W_q + b_q              (bf16)
    gemm_wmma_kernel<128, 128, 4, 2><<<dim3(H_ / 128, B_ / 128), 256, 0, stream>>>(
        combOut, OGK_, WtQ, b_q, B_, H_, H_,
        2, nullptr, 0, queryB, H_, nullptr, 0);

    // 5) logits = query @ W_sel + b_sel         (f32, N=64)
    gemm_wmma_kernel<128, 64, 8, 1><<<dim3(S_ / 64, B_ / 128), 256, 0, stream>>>(
        queryB, H_, WtSel, b_sel, B_, S_, H_,
        0, logits, S_, nullptr, 0, nullptr, 0);

    // 6) argmax + gathered einsums -> m_t bf16 into combOut[:, H:]
    select_einsum_kernel<<<B_, 256, 0, stream>>>(
        logits, hintF, Wm_enc, theta, Wm_dec, b_m, combOut + H_, OGK_);

    // 7) goh = [h_int||m_t] @ W_og + b_og ; h_next = tanh(goh[:, :H]), y_t = goh[:, H:]
    gemm_wmma_kernel<128, 128, 4, 2><<<dim3(OGN_ / 128, B_ / 128), 256, 0, stream>>>(
        combOut, OGK_, WtOg, b_og, B_, OGN_, OGK_,
        3, h_next, H_, nullptr, 0, y_t, H_);
}
